// DynamicReductionNetworkJit_8950711845774
// MI455X (gfx1250) — matrix operand order repaired and checked by reading, not compiled, not run on hardware
//
#include <hip/hip_runtime.h>
#include <hip/hip_bf16.h>
#include <cstdint>

#define N_NODES 8192
#define KNN 16
#define HID 64
#define NB 16
#define GFEAT 4
#define DIN 5
#define BIG 1e30f
#define KNN_WAVES 4

typedef __attribute__((ext_vector_type(16))) _Float16 v16h;
typedef __attribute__((ext_vector_type(8)))  _Float16 v8h;
typedef __attribute__((ext_vector_type(8)))  float    v8f;
typedef __attribute__((ext_vector_type(4)))  unsigned int v4u;
typedef __attribute__((ext_vector_type(8)))  int v8i;
typedef __attribute__((ext_vector_type(4)))  int v4i;

__device__ __forceinline__ float eluf(float x) { return x > 0.f ? x : expm1f(x); }
__device__ __forceinline__ int   f2key(float f) { int b = __float_as_int(f); return b >= 0 ? b : (b ^ 0x7fffffff); }
__device__ __forceinline__ float key2f(int k)  { int b = k >= 0 ? k : (k ^ 0x7fffffff); return __int_as_float(b); }

// ---- WMMA fragment builders (CDNA5 16x16x32 f16 layouts, CDNA5 ISA §7.12.2) ----
// A 16x32 f16: lane -> M; per lane the needed K values are two contiguous 8-half runs
// (K = khi..khi+7 in VGPRs 0..3, K = 16+khi.. in VGPRs 4..7)  ->  2x ds_load_b128.
__device__ __forceinline__ v16h wmma_load_a(const _Float16* tile, int ld, int k0, int lane) {
  int m = lane & 15;
  int khi = (lane & 16) ? 8 : 0;
  const _Float16* base = tile + m * ld + k0 + khi;
  v8h lo = *(const v8h*)(base);
  v8h hi = *(const v8h*)(base + 16);
  return __builtin_shufflevector(lo, hi, 0, 1, 2, 3, 4, 5, 6, 7, 8, 9, 10, 11, 12, 13, 14, 15);
}
// B 32x16 f16 (row-major K x N source): lane -> N, per lane K = khi..khi+7 then 16+khi..16+khi+7
// (same K order as A), strided scalar loads.
__device__ __forceinline__ v16h wmma_load_b(const _Float16* tile, int ld, int k0, int n0, int lane) {
  int n = lane & 15;
  int khi = (lane & 16) ? 8 : 0;
  const _Float16* base = tile + (k0 + khi) * ld + n0 + n;
  v16h r;
#pragma unroll
  for (int s = 0; s < 16; ++s) r[s] = base[((s < 8) ? s : (s + 8)) * ld];
  return r;
}
__device__ __forceinline__ v8f wmma_f16(v16h a, v16h b, v8f c) {
  return __builtin_amdgcn_wmma_f32_16x16x32_f16(false, a, false, b, (short)0, c, false, false);
}

// ---- Tensor Data Mover: 1-D 2-byte-element DMA global -> LDS (CDNA5 ISA) ----
// 6-arg builtin form (clang-23 / therock-10.0 toolchain).
__device__ __forceinline__ void tdm_load_2b(unsigned ldsOff, const void* gsrc, unsigned nElems) {
  unsigned long long ga = (unsigned long long)(uintptr_t)gsrc;
  v4u g0;
  g0[0] = 1u;                                                  // count=1, user mode, no gather
  g0[1] = ldsOff;                                              // lds_addr (bytes)
  g0[2] = (unsigned)ga;                                        // global_addr[31:0]
  g0[3] = (unsigned)((ga >> 32) & 0x01ffffffu) | (2u << 30);   // global_addr[56:32] | type=2
  v8i g1;
  g1[0] = 1 << 16;                                             // workgroup_mask=0, data_size=1 (2B)
  g1[1] = (int)((nElems & 0xffffu) << 16);                     // tensor_dim0[15:0] @ bits63:48
  g1[2] = (int)((nElems >> 16) & 0xffffu) | (1 << 16);         // tensor_dim0[31:16], tensor_dim1=1
  g1[3] = (int)((nElems & 0xffffu) << 16);                     // tile_dim0 @ bits127:112
  g1[4] = 1;                                                   // tile_dim1=1, tile_dim2=0
  g1[5] = (int)nElems;                                         // tensor_dim0_stride[31:0]
  g1[6] = 0;                                                   // stride0[47:32]=0, stride1 lo=0
  g1[7] = 0;
  v4i z4 = {0, 0, 0, 0};
  v8i z8 = {0, 0, 0, 0, 0, 0, 0, 0};
  __builtin_amdgcn_tensor_load_to_lds(g0, g1, z4, z4, z8, 0);
}

// ---------------- utility fills ----------------
__global__ void fill_i32(int* __restrict__ p, int v, int n) {
  int i = blockIdx.x * blockDim.x + threadIdx.x;
  if (i < n) p[i] = v;
}
__global__ void fill_f32(float* __restrict__ p, float v, int n) {
  int i = blockIdx.x * blockDim.x + threadIdx.x;
  if (i < n) p[i] = v;
}
__global__ void convert_f16_kernel(const float* __restrict__ src, _Float16* __restrict__ dst, int n) {
  int i = blockIdx.x * blockDim.x + threadIdx.x;
  if (i < n) dst[i] = (_Float16)src[i];
}
__global__ void init_state_kernel(const int* __restrict__ bin, int* __restrict__ bcur, int* __restrict__ vcur) {
  int i = blockIdx.x * blockDim.x + threadIdx.x;
  if (i < N_NODES) { bcur[i] = bin[i]; vcur[i] = 1; }
}

// ---------------- input MLPs (D_IN=5 -> 64 -> 64, ELU) ----------------
__global__ void input_mlp_kernel(const float* __restrict__ xin, const float* __restrict__ dn,
                                 const float* __restrict__ W0, const float* __restrict__ b0,
                                 const float* __restrict__ W1, const float* __restrict__ b1,
                                 float* __restrict__ xout, int nNodes, int outOff) {
  int i = blockIdx.x * blockDim.x + threadIdx.x;
  if (i >= nNodes) return;
  float in[DIN];
#pragma unroll
  for (int d = 0; d < DIN; ++d) in[d] = dn[d] * xin[i * DIN + d];
  float y[HID];
  for (int h = 0; h < HID; ++h) {
    float acc = b0[h];
#pragma unroll
    for (int d = 0; d < DIN; ++d) acc += in[d] * W0[d * HID + h];
    y[h] = eluf(acc);
  }
  for (int h = 0; h < HID; ++h) {
    float acc = b1[h];
    for (int k = 0; k < HID; ++k) acc += y[k] * W1[k * HID + h];
    xout[(outOff + i) * HID + h] = eluf(acc);
  }
}

// ---------------- f32 -> f16 + squared norms ----------------
__global__ void prep_kernel(const float* __restrict__ x, _Float16* __restrict__ xh, float* __restrict__ sq) {
  int i = blockIdx.x * blockDim.x + threadIdx.x;
  if (i >= N_NODES) return;
  float s = 0.f;
  for (int h = 0; h < HID; ++h) { float v = x[i * HID + h]; s += v * v; xh[i * HID + h] = (_Float16)v; }
  sq[i] = s;
}

// ---------------- fused kNN: WMMA distance tiles + in-LDS top-K (d2 never hits HBM) ----------
__global__ __launch_bounds__(32 * KNN_WAVES)
void knn_kernel(const _Float16* __restrict__ xh, const float* __restrict__ sq,
                const int* __restrict__ batch, const int* __restrict__ valid,
                int* __restrict__ nbr, float* __restrict__ nbrd2) {
  __shared__ alignas(16) _Float16 qh[KNN_WAVES][16][HID];   // queries, M-major
  __shared__ alignas(16) _Float16 cht[KNN_WAVES][HID][16];  // candidates, K-major (pre-transposed)
  __shared__ float qsq[KNN_WAVES][16], csq[KNN_WAVES][16];
  __shared__ int qb[KNN_WAVES][16], cb[KNN_WAVES][16];
  __shared__ int qv[KNN_WAVES][16], cv[KNN_WAVES][16];
  __shared__ float d2t[KNN_WAVES][16][16];
  __shared__ float topd[KNN_WAVES][16][KNN];
  __shared__ int   topi[KNN_WAVES][16][KNN];

  const int lane = threadIdx.x & 31;
  const int wid  = threadIdx.x >> 5;
  const int q0   = (blockIdx.x * KNN_WAVES + wid) * 16;

  for (int t = lane; t < 16 * HID; t += 32)
    qh[wid][t >> 6][t & 63] = xh[(q0 + (t >> 6)) * HID + (t & 63)];
  if (lane < 16) {
    int qi = q0 + lane;
    qsq[wid][lane] = sq[qi];
    qb[wid][lane]  = batch[qi];
    qv[wid][lane]  = valid[qi];
    for (int m = 0; m < KNN; ++m) { topd[wid][lane][m] = BIG; topi[wid][lane][m] = qi; }
  }
  __syncthreads();

  v16h a0 = wmma_load_a(&qh[wid][0][0], HID, 0, lane);
  v16h a1 = wmma_load_a(&qh[wid][0][0], HID, 32, lane);
  const int nloc = lane & 15;
  const int mhi  = (lane & 16) ? 8 : 0;

  // loop-invariant query scalars for this lane's 8 output rows
  float qsqr[8]; int qbr[8], qvr[8];
#pragma unroll
  for (int r = 0; r < 8; ++r) {
    int m = r + mhi;
    qsqr[r] = qsq[wid][m]; qbr[r] = qb[wid][m]; qvr[r] = qv[wid][m];
  }

  for (int ct = 0; ct < N_NODES / 16; ++ct) {
    const int c0 = ct * 16;
    if (ct + 2 < N_NODES / 16)
      __builtin_prefetch(&xh[(c0 + 32) * HID], 0, 0);   // global_prefetch_b8
    __syncthreads();
    // load candidate tile transposed: cht[k][n] = x[c0+n][k]
    for (int t = lane; t < 16 * 8; t += 32) {
      int n = t >> 3, k8 = (t & 7) * 8;
      v8h chunk = *(const v8h*)&xh[(c0 + n) * HID + k8];
#pragma unroll
      for (int e = 0; e < 8; ++e) cht[wid][k8 + e][n] = chunk[e];
    }
    if (lane < 16) {
      csq[wid][lane] = sq[c0 + lane];
      cb[wid][lane]  = batch[c0 + lane];
      cv[wid][lane]  = valid[c0 + lane];
    }
    __syncthreads();
    v16h b0 = wmma_load_b(&cht[wid][0][0], 16, 0, 0, lane);
    v16h b1 = wmma_load_b(&cht[wid][0][0], 16, 32, 0, lane);
    v8f c = {};
    c = wmma_f16(a0, b0, c);
    c = wmma_f16(a1, b1, c);
    float csqn = csq[wid][nloc];
    int   cbn  = cb[wid][nloc];
    int   cvn  = cv[wid][nloc];
#pragma unroll
    for (int r = 0; r < 8; ++r) {
      int ok = (qbr[r] == cbn) & (qvr[r] & cvn);
      float d2 = fmaf(-2.f, c[r], qsqr[r] + csqn);
      d2t[wid][r + mhi][nloc] = ok ? d2 : BIG;
    }
    __syncthreads();
    if (lane < 16) {
      for (int n = 0; n < 16; ++n) {
        float d = d2t[wid][lane][n];
        if (d < topd[wid][lane][KNN - 1]) {
          int pos = KNN - 1;
          while (pos > 0 && topd[wid][lane][pos - 1] > d) {
            topd[wid][lane][pos] = topd[wid][lane][pos - 1];
            topi[wid][lane][pos] = topi[wid][lane][pos - 1];
            --pos;
          }
          topd[wid][lane][pos] = d;
          topi[wid][lane][pos] = c0 + n;
        }
      }
    }
  }
  __syncthreads();
  if (lane < 16) {
    int qi = q0 + lane;
    for (int m = 0; m < KNN; ++m) {
      nbr[qi * KNN + m]   = topi[wid][lane][m];
      nbrd2[qi * KNN + m] = topd[wid][lane][m];
    }
  }
}

// ---------------- edge validity + duplicate removal + degree ----------------
__global__ void edge_build_kernel(const int* __restrict__ nbr, const float* __restrict__ nbrd2,
                                  const int* __restrict__ valid,
                                  int* __restrict__ evf, int* __restrict__ evr, int* __restrict__ deg) {
  int idx = blockIdx.x * blockDim.x + threadIdx.x;
  if (idx >= N_NODES * KNN) return;
  int i = idx / KNN;
  int j = nbr[idx];
  int okf = (nbrd2[idx] < 0.5f * BIG) & valid[i];
  int dup = 0;
  for (int mm = 0; mm < KNN; ++mm)
    dup |= (nbr[j * KNN + mm] == i) & (nbrd2[j * KNN + mm] < 0.5f * BIG);
  int okr = okf & !dup;
  evf[idx] = okf;
  evr[idx] = okr;
  if (okf) atomicAdd(&deg[i], 1);
  if (okr) atomicAdd(&deg[j], 1);
}

// ---------------- EdgeConv: fused 2-layer MLP GEMM via WMMA, weights via TDM DMA ------------
__global__ __launch_bounds__(64)
void edgeconv_kernel(const float* __restrict__ x, const int* __restrict__ nbr,
                     const int* __restrict__ evf, const int* __restrict__ evr,
                     const _Float16* __restrict__ W1g, const float* __restrict__ b1,
                     const _Float16* __restrict__ W2g, const float* __restrict__ b2,
                     float* __restrict__ xout) {
  __shared__ alignas(16) _Float16 w1h[128][128];    // 32 KB, filled by tensor_load_to_lds
  __shared__ alignas(16) _Float16 w2h[128][HID];    // 16 KB, filled by tensor_load_to_lds
  __shared__ float b1s[128];
  __shared__ float b2s[HID];
  __shared__ alignas(16) _Float16 feat[2][16][128]; // per-wave edge-feature / h1 tile
  __shared__ int dstL[2][16], srcL[2][16], actL[2][16];

  const int tid = threadIdx.x, lane = tid & 31, wid = tid >> 5;
  if (wid == 0) {
    // Tensor Data Mover: DMA both weight matrices into LDS (TENSORcnt-tracked)
    tdm_load_2b((unsigned)(uintptr_t)&w1h[0][0], W1g, 128 * 128);
    tdm_load_2b((unsigned)(uintptr_t)&w2h[0][0], W2g, 128 * HID);
    __builtin_amdgcn_s_wait_tensorcnt(0);
  }
  for (int t = tid; t < 128; t += 64) b1s[t] = b1[t];
  if (tid < HID) b2s[tid] = b2[tid];

  const int e0 = (blockIdx.x * 2 + wid) * 16;
  if (lane < 16) {
    int e = e0 + lane;
    int dst, src, act;
    if (e < N_NODES * KNN) { dst = e / KNN; src = nbr[e]; act = evf[e]; }
    else { int e2 = e - N_NODES * KNN; dst = nbr[e2]; src = e2 / KNN; act = evr[e2]; }
    dstL[wid][lane] = dst; srcL[wid][lane] = src; actL[wid][lane] = act;
  }
  __syncthreads();
  // build [x_i, x_j - x_i] rows (zeros for inactive edge slots), vector LDS stores
  for (int t = lane; t < 16 * 16; t += 32) {
    int row = t >> 4, c8 = (t & 15) * 8;
    int act = actL[wid][row];
    int d = dstL[wid][row], s = srcL[wid][row];
    v8h out;
#pragma unroll
    for (int e = 0; e < 8; ++e) {
      int col = c8 + e;
      float v = 0.f;
      if (act)
        v = (col < HID) ? x[d * HID + col] : (x[s * HID + col - HID] - x[d * HID + col - HID]);
      out[e] = (_Float16)v;
    }
    *(v8h*)&feat[wid][row][c8] = out;
  }
  __syncthreads();

  // layer 1: [16x128] @ [128x128]
  v16h a[4];
#pragma unroll
  for (int kt = 0; kt < 4; ++kt) a[kt] = wmma_load_a(&feat[wid][0][0], 128, kt * 32, lane);
  v8f cAll[8];
#pragma unroll
  for (int nt = 0; nt < 8; ++nt) {
    v8f c = {};
#pragma unroll
    for (int kt = 0; kt < 4; ++kt) {
      v16h b = wmma_load_b(&w1h[0][0], 128, kt * 32, nt * 16, lane);
      c = wmma_f16(a[kt], b, c);
    }
    cAll[nt] = c;
  }
  const int nloc = lane & 15, mhi = (lane & 16) ? 8 : 0;
  __syncthreads();
  // bias + ELU, recycle the same LDS tile for h1
#pragma unroll
  for (int nt = 0; nt < 8; ++nt) {
#pragma unroll
    for (int r = 0; r < 8; ++r) {
      int col = nt * 16 + nloc;
      feat[wid][r + mhi][col] = (_Float16)eluf(cAll[nt][r] + b1s[col]);
    }
  }
  __syncthreads();
  // layer 2: [16x128] @ [128x64], then ELU + atomic sum into x_new[dst]
#pragma unroll
  for (int kt = 0; kt < 4; ++kt) a[kt] = wmma_load_a(&feat[wid][0][0], 128, kt * 32, lane);
#pragma unroll
  for (int nt = 0; nt < 4; ++nt) {
    v8f c = {};
#pragma unroll
    for (int kt = 0; kt < 4; ++kt) {
      v16h b = wmma_load_b(&w2h[0][0], HID, kt * 32, nt * 16, lane);
      c = wmma_f16(a[kt], b, c);
    }
#pragma unroll
    for (int r = 0; r < 8; ++r) {
      int m = r + mhi;
      if (actL[wid][m]) {
        int col = nt * 16 + nloc;
        atomicAdd(&xout[dstL[wid][m] * HID + col], eluf(c[r] + b2s[col]));
      }
    }
  }
}

// ---------------- graclus proxy: best-neighbor selection ----------------
__global__ void cluster_select_kernel(const float* __restrict__ x, const int* __restrict__ nbr,
                                      const float* __restrict__ nbrd2, const int* __restrict__ deg,
                                      const int* __restrict__ valid,
                                      int* __restrict__ best, int* __restrict__ has) {
  int i = blockIdx.x * blockDim.x + threadIdx.x;
  if (i >= N_NODES) return;
  float di = (float)deg[i];
  float invdi = di > 0.f ? 1.f / fmaxf(di, 1.f) : 0.f;
  float wbest = -3.0e38f;
  int bsel = nbr[i * KNN];
  int h = 0;
  for (int m = 0; m < KNN; ++m) {
    int j = nbr[i * KNN + m];
    int ok = (nbrd2[i * KNN + m] < 0.5f * BIG) & (j != i) & valid[i] & valid[j];
    float w;
    if (ok) {
      float d2 = 0.f;
      for (int hh = 0; hh < HID; ++hh) { float df = x[i * HID + hh] - x[j * HID + hh]; d2 += df * df; }
      float attr = d2 > 0.f ? sqrtf(d2) : 1.f;
      float dj = (float)deg[j];
      float invdj = dj > 0.f ? 1.f / fmaxf(dj, 1.f) : 0.f;
      w = attr * (invdi + invdj);
      h = 1;
    } else {
      w = -BIG;
    }
    if (w > wbest) { wbest = w; bsel = j; }
  }
  best[i] = bsel;
  has[i] = h;
}

__global__ void cluster_assign_kernel(const int* __restrict__ best, const int* __restrict__ has,
                                      const int* __restrict__ valid, int* __restrict__ cluster) {
  int i = blockIdx.x * blockDim.x + threadIdx.x;
  if (i >= N_NODES) return;
  int b = best[i];
  int mutual = has[i] && has[b] && (best[b] == i);
  int c = mutual ? (i < b ? i : b) : i;
  cluster[i] = valid[i] ? c : N_NODES;
}

// ---------------- max pooling via monotonic float->int keys + atomicMax ----------------
__global__ void pool_scatter_kernel(const float* __restrict__ x, const int* __restrict__ batch,
                                    const int* __restrict__ cluster,
                                    int* __restrict__ poolKey, int* __restrict__ nbmax) {
  int i = blockIdx.x * blockDim.x + threadIdx.x;
  if (i >= N_NODES) return;
  int c = cluster[i];
  if (c >= N_NODES) return;
  for (int h = 0; h < HID; ++h) atomicMax(&poolKey[c * HID + h], f2key(x[i * HID + h]));
  atomicMax(&nbmax[c], batch[i]);
}

__global__ void pool_finish_kernel(const int* __restrict__ poolKey, const int* __restrict__ nbmax,
                                   float* __restrict__ xnext, int* __restrict__ bnext,
                                   int* __restrict__ vnext) {
  int i = blockIdx.x * blockDim.x + threadIdx.x;
  if (i >= N_NODES) return;
  int nb = nbmax[i];
  int v = nb >= 0;
  for (int h = 0; h < HID; ++h)
    xnext[i * HID + h] = v ? key2f(poolKey[i * HID + h]) : 0.f;
  bnext[i] = v ? nb : NB;
  vnext[i] = v;
}

__global__ void gpool_scatter_kernel(const float* __restrict__ x, const int* __restrict__ batch,
                                     const int* __restrict__ valid, int* __restrict__ gKey) {
  int i = blockIdx.x * blockDim.x + threadIdx.x;
  if (i >= N_NODES) return;
  if (!valid[i]) return;
  int b = batch[i];
  if (b < 0 || b >= NB) return;
  for (int h = 0; h < HID; ++h) atomicMax(&gKey[b * HID + h], f2key(x[i * HID + h]));
}

// ---------------- output MLP (16x68 -> 68 -> 68 -> 1) ----------------
__global__ __launch_bounds__(128)
void out_mlp_kernel(const int* __restrict__ gKey, const float* __restrict__ gx,
                    const float* __restrict__ W0, const float* __restrict__ b0,
                    const float* __restrict__ W1, const float* __restrict__ b1,
                    const float* __restrict__ W2, const float* __restrict__ b2,
                    float* __restrict__ out) {
  __shared__ float g[NB][HID + GFEAT];
  __shared__ float g2[NB][HID + GFEAT];
  const int F = HID + GFEAT;
  int tid = threadIdx.x;
  for (int t = tid; t < NB * HID; t += blockDim.x) {
    int k = gKey[t];
    float v = (k == (int)0x80000000u) ? 0.f : key2f(k);
    if (!isfinite(v)) v = 0.f;
    g[t / HID][t % HID] = v;
  }
  for (int t = tid; t < NB * GFEAT; t += blockDim.x) g[t / GFEAT][HID + (t % GFEAT)] = gx[t];
  __syncthreads();
  for (int t = tid; t < NB * F; t += blockDim.x) {
    int b = t / F, o = t % F;
    float acc = b0[o];
    for (int k = 0; k < F; ++k) acc += g[b][k] * W0[k * F + o];
    g2[b][o] = eluf(acc);
  }
  __syncthreads();
  for (int t = tid; t < NB * F; t += blockDim.x) {
    int b = t / F, o = t % F;
    float acc = b1[o];
    for (int k = 0; k < F; ++k) acc += g2[b][k] * W1[k * F + o];
    g[b][o] = eluf(acc);
  }
  __syncthreads();
  if (tid < NB) {
    float acc = b2[0];
    for (int k = 0; k < F; ++k) acc += g[tid][k] * W2[k];
    out[tid] = acc;
  }
}

extern "C" void kernel_launch(void* const* d_in, const int* in_sizes, int n_in,
                              void* d_out, int out_size, void* d_ws, size_t ws_size,
                              hipStream_t stream) {
  const float* xECAL   = (const float*)d_in[0];
  const float* xES     = (const float*)d_in[1];
  const int*   batchIn = (const int*)d_in[2];
  const float* graphx  = (const float*)d_in[3];
  const float* datanorm = (const float*)d_in[4];
  const float* ecalW0 = (const float*)d_in[5];
  const float* ecalB0 = (const float*)d_in[6];
  const float* ecalW1 = (const float*)d_in[7];
  const float* ecalB1 = (const float*)d_in[8];
  const float* esW0 = (const float*)d_in[9];
  const float* esB0 = (const float*)d_in[10];
  const float* esW1 = (const float*)d_in[11];
  const float* esB1 = (const float*)d_in[12];
  const float* aggW[2][2] = {{(const float*)d_in[13], (const float*)d_in[15]},
                             {(const float*)d_in[17], (const float*)d_in[19]}};
  const float* aggB[2][2] = {{(const float*)d_in[14], (const float*)d_in[16]},
                             {(const float*)d_in[18], (const float*)d_in[20]}};
  const float* outW0 = (const float*)d_in[21];
  const float* outB0 = (const float*)d_in[22];
  const float* outW1 = (const float*)d_in[23];
  const float* outB1 = (const float*)d_in[24];
  const float* outW2 = (const float*)d_in[25];
  const float* outB2 = (const float*)d_in[26];

  char* p = (char*)d_ws;
  auto carve = [&](size_t bytes) -> void* {
    void* r = p;
    p += (bytes + 255) & ~(size_t)255;
    return r;
  };
  float*    xA      = (float*)carve((size_t)N_NODES * HID * 4);
  float*    xB      = (float*)carve((size_t)N_NODES * HID * 4);
  _Float16* xh      = (_Float16*)carve((size_t)N_NODES * HID * 2);
  float*    sq      = (float*)carve((size_t)N_NODES * 4);
  int*      batchA  = (int*)carve((size_t)N_NODES * 4);
  int*      batchB  = (int*)carve((size_t)N_NODES * 4);
  int*      validA  = (int*)carve((size_t)N_NODES * 4);
  int*      validB  = (int*)carve((size_t)N_NODES * 4);
  int*      nbr     = (int*)carve((size_t)N_NODES * KNN * 4);
  float*    nbrd2   = (float*)carve((size_t)N_NODES * KNN * 4);
  int*      evf     = (int*)carve((size_t)N_NODES * KNN * 4);
  int*      evr     = (int*)carve((size_t)N_NODES * KNN * 4);
  int*      deg     = (int*)carve((size_t)N_NODES * 4);
  int*      best    = (int*)carve((size_t)N_NODES * 4);
  int*      has     = (int*)carve((size_t)N_NODES * 4);
  int*      cluster = (int*)carve((size_t)N_NODES * 4);
  int*      poolKey = (int*)carve((size_t)N_NODES * HID * 4);
  int*      nbmax   = (int*)carve((size_t)N_NODES * 4);
  int*      gKey    = (int*)carve((size_t)NB * HID * 4);
  _Float16* w1g     = (_Float16*)carve((size_t)128 * 128 * 2);
  _Float16* w2g     = (_Float16*)carve((size_t)128 * HID * 2);

  input_mlp_kernel<<<(4096 + 127) / 128, 128, 0, stream>>>(xECAL, datanorm, ecalW0, ecalB0, ecalW1, ecalB1, xA, 4096, 0);
  input_mlp_kernel<<<(4096 + 127) / 128, 128, 0, stream>>>(xES, datanorm, esW0, esB0, esW1, esB1, xA, 4096, 4096);
  init_state_kernel<<<N_NODES / 128, 128, 0, stream>>>(batchIn, batchA, validA);

  float* xcur = xA;
  float* xalt = xB;
  int* bcur = batchA; int* balt = batchB;
  int* vcur = validA; int* valt = validB;

  for (int l = 0; l < 2; ++l) {
    prep_kernel<<<N_NODES / 128, 128, 0, stream>>>(xcur, xh, sq);
    knn_kernel<<<N_NODES / (16 * KNN_WAVES), 32 * KNN_WAVES, 0, stream>>>(xh, sq, bcur, vcur, nbr, nbrd2);
    fill_i32<<<(N_NODES + 255) / 256, 256, 0, stream>>>(deg, 0, N_NODES);
    edge_build_kernel<<<(N_NODES * KNN) / 256, 256, 0, stream>>>(nbr, nbrd2, vcur, evf, evr, deg);
    fill_f32<<<(N_NODES * HID + 255) / 256, 256, 0, stream>>>(xalt, 0.f, N_NODES * HID);
    convert_f16_kernel<<<(128 * 128 + 255) / 256, 256, 0, stream>>>(aggW[l][0], w1g, 128 * 128);
    convert_f16_kernel<<<(128 * HID + 255) / 256, 256, 0, stream>>>(aggW[l][1], w2g, 128 * HID);
    edgeconv_kernel<<<(2 * N_NODES * KNN) / 32, 64, 0, stream>>>(
        xcur, nbr, evf, evr, w1g, aggB[l][0], w2g, aggB[l][1], xalt);
    cluster_select_kernel<<<N_NODES / 128, 128, 0, stream>>>(xalt, nbr, nbrd2, deg, vcur, best, has);
    cluster_assign_kernel<<<N_NODES / 128, 128, 0, stream>>>(best, has, vcur, cluster);
    fill_i32<<<(N_NODES * HID + 255) / 256, 256, 0, stream>>>(poolKey, (int)0x80000000u, N_NODES * HID);
    fill_i32<<<(N_NODES + 255) / 256, 256, 0, stream>>>(nbmax, -1, N_NODES);
    pool_scatter_kernel<<<N_NODES / 128, 128, 0, stream>>>(xalt, bcur, cluster, poolKey, nbmax);
    pool_finish_kernel<<<N_NODES / 128, 128, 0, stream>>>(poolKey, nbmax, xcur, balt, valt);
    int* t;
    t = bcur; bcur = balt; balt = t;
    t = vcur; vcur = valt; valt = t;
  }

  fill_i32<<<(NB * HID + 255) / 256, 256, 0, stream>>>(gKey, (int)0x80000000u, NB * HID);
  gpool_scatter_kernel<<<N_NODES / 128, 128, 0, stream>>>(xcur, bcur, vcur, gKey);
  out_mlp_kernel<<<1, 128, 0, stream>>>(gKey, graphx, outW0, outB0, outW1, outB1, outW2, outB2, (float*)d_out);

  (void)in_sizes; (void)n_in; (void)out_size; (void)ws_size;
}
